// GCL_67018669687401
// MI455X (gfx1250) — matrix operand order repaired and checked by reading, not compiled, not run on hardware
//
#include <hip/hip_runtime.h>

// ---------------------------------------------------------------------------
// GCL (EGNN graph conv layer) for gfx1250, fused, bf16 WMMA w/ fp32 accum.
// N=50000 nodes, E=800000 edges, D=H=128.
// Edge kernel: 32 edges (2 M-tiles) per wave so each B-fragment load feeds
// two WMMAs (vmem:wmma ratio ~1.2 instead of ~2.2).
// ---------------------------------------------------------------------------

#define N_NODES 50000
#define N_EDGES 800000
#define DF      128

typedef __bf16 v16bf __attribute__((ext_vector_type(16)));
typedef __bf16 v8bf  __attribute__((ext_vector_type(8)));
typedef float  v8f   __attribute__((ext_vector_type(8)));

__device__ __forceinline__ float silu_f(float x) {
    return x / (1.0f + __expf(-x));
}

// ---------------------------------------------------------------------------
// Prep: h -> bf16 copy, zero agg (same element count N*128 for both).
// ---------------------------------------------------------------------------
__global__ void gcl_prep_h(const float* __restrict__ h,
                           __bf16* __restrict__ hb,
                           float* __restrict__ agg, int total) {
    int idx = blockIdx.x * blockDim.x + threadIdx.x;
    if (idx < total) {
        hb[idx]  = (__bf16)h[idx];
        agg[idx] = 0.0f;
    }
}

// ---------------------------------------------------------------------------
// Repack a [K,128] fp32 weight matrix into bf16 WMMA B-fragments.
// Fragment (kt, nt) covers K rows [kt*32, kt*32+32), N cols [nt*16, nt*16+16).
// In-register B layout (ISA 7.12.2, 16-bit B 32x16):
//   lane L in [0,16):  col N = L,     holds K = kt*32 + 0..7, 16..23   (16 bf16)
//   lane L in [16,32): col N = L-16,  holds K = kt*32 + 8..15, 24..31  (16 bf16)
// Memory order: frag-major, then lane-major: each lane reads one 32B v16bf.
// ---------------------------------------------------------------------------
__global__ void gcl_pack_w(const float* __restrict__ W,
                           __bf16* __restrict__ out, int kt_count) {
    int o = blockIdx.x * blockDim.x + threadIdx.x;
    int total = kt_count * 8 * 512;          // frags * 512 elems
    if (o >= total) return;
    int i    = o & 15;
    int L    = (o >> 4) & 31;
    int frag = o >> 9;
    int nt   = frag & 7;
    int kt   = frag >> 3;
    int n    = nt * 16 + (L & 15);
    int k    = kt * 32 + (L >> 4) * 8 + (i & 7) + (i >> 3) * 16;
    out[o] = (__bf16)W[k * DF + n];
}

// ---------------------------------------------------------------------------
// Edge kernel: one wave32 per 32-edge tile (two 16-row M-tiles t=0,1).
//   e_in = concat(h[row], h[col])              [32, 256] (bf16 gather)
//   e1   = silu(e_in @ eW1 + eb1)              [32, 128]
//   e2   = silu(e1 @ eW2 + eb2)                [32, 128]
//   agg[row] += e2                             (atomic f32 scatter)
// A-fragment layout (ISA 7.12.2, 16-bit A 16x32), half = L>>4, r = L&15:
//   a[0..7]  = A[r][half*8 + 0..7]
//   a[8..15] = A[r][16 + half*8 + 0..7]
// ---------------------------------------------------------------------------
__global__ __launch_bounds__(32)
void gcl_edge_kernel(const __bf16* __restrict__ hb,
                     const int*    __restrict__ rows,
                     const int*    __restrict__ cols,
                     const __bf16* __restrict__ W1p, const float* __restrict__ b1,
                     const __bf16* __restrict__ W2p, const float* __restrict__ b2,
                     float* __restrict__ agg) {
    const int L    = threadIdx.x;
    const int r    = L & 15;
    const int half = L >> 4;
    const int e0   = blockIdx.x * 32;

    int nr[2], nc[2];
#pragma unroll
    for (int t = 0; t < 2; ++t) {
        nr[t] = rows[e0 + t * 16 + r];
        nc[t] = cols[e0 + t * 16 + r];
    }

    __shared__ __bf16 sE[2][16 * DF];   // 8KB: e1 tiles, row-major [16][128]

    // ---------------- layer 1: e_in[32,256] @ eW1[256,128] ----------------
    v8f acc[2][8];
#pragma unroll
    for (int t = 0; t < 2; ++t)
#pragma unroll
        for (int nt = 0; nt < 8; ++nt)
#pragma unroll
            for (int j = 0; j < 8; ++j) acc[t][nt][j] = 0.0f;

#pragma unroll
    for (int kk = 0; kk < 8; ++kk) {            // K = kk*32 .. kk*32+31
        const int cb = (kk * 32) & 127;         // cols 0-127 row side, 128-255 col side
        v16bf a[2];
#pragma unroll
        for (int t = 0; t < 2; ++t) {
            const int node = (kk < 4) ? nr[t] : nc[t];
            const __bf16* src = hb + node * DF + cb + half * 8;
            v8bf lo = *(const v8bf*)(src);
            v8bf hi = *(const v8bf*)(src + 16);
#pragma unroll
            for (int i = 0; i < 8; ++i) { a[t][i] = lo[i]; a[t][8 + i] = hi[i]; }
        }
#pragma unroll
        for (int nt = 0; nt < 8; ++nt) {
            const v16bf* bp = (const v16bf*)(W1p + ((kk * 8 + nt) << 9));
            v16bf b = bp[L];
#pragma unroll
            for (int t = 0; t < 2; ++t)
                acc[t][nt] = __builtin_amdgcn_wmma_f32_16x16x32_bf16(
                    false, a[t], false, b, (short)0, acc[t][nt], false, false);
        }
    }

    // bias + SiLU, write e1 to LDS row-major (D-layout -> row-major transpose)
#pragma unroll
    for (int nt = 0; nt < 8; ++nt) {
        const int colN = nt * 16 + r;           // D-layout: N = L&15
        const float bias = b1[colN];
#pragma unroll
        for (int t = 0; t < 2; ++t)
#pragma unroll
            for (int v = 0; v < 8; ++v) {
                float x = silu_f(acc[t][nt][v] + bias);
                sE[t][(v + 8 * half) * DF + colN] = (__bf16)x;  // row M = v+8*half
            }
    }
    __syncthreads();

    // ---------------- layer 2: e1[32,128] @ eW2[128,128] -------------------
    v8f acc2[2][8];
#pragma unroll
    for (int t = 0; t < 2; ++t)
#pragma unroll
        for (int nt = 0; nt < 8; ++nt)
#pragma unroll
            for (int j = 0; j < 8; ++j) acc2[t][nt][j] = 0.0f;

#pragma unroll
    for (int kk = 0; kk < 4; ++kk) {
        v16bf a[2];
#pragma unroll
        for (int t = 0; t < 2; ++t) {
            const __bf16* src = &sE[t][r * DF + kk * 32 + half * 8];
            v8bf lo = *(const v8bf*)(src);
            v8bf hi = *(const v8bf*)(src + 16);
#pragma unroll
            for (int i = 0; i < 8; ++i) { a[t][i] = lo[i]; a[t][8 + i] = hi[i]; }
        }
#pragma unroll
        for (int nt = 0; nt < 8; ++nt) {
            const v16bf* bp = (const v16bf*)(W2p + ((kk * 8 + nt) << 9));
            v16bf b = bp[L];
#pragma unroll
            for (int t = 0; t < 2; ++t)
                acc2[t][nt] = __builtin_amdgcn_wmma_f32_16x16x32_bf16(
                    false, a[t], false, b, (short)0, acc2[t][nt], false, false);
        }
    }

    // bias + SiLU + atomic scatter-add into agg[row]
    int er[2][8];
#pragma unroll
    for (int t = 0; t < 2; ++t)
#pragma unroll
        for (int v = 0; v < 8; ++v)
            er[t][v] = rows[e0 + t * 16 + 8 * half + v];

#pragma unroll
    for (int nt = 0; nt < 8; ++nt) {
        const int colN = nt * 16 + r;
        const float bias = b2[colN];
#pragma unroll
        for (int t = 0; t < 2; ++t)
#pragma unroll
            for (int v = 0; v < 8; ++v) {
                float x = silu_f(acc2[t][nt][v] + bias);
                atomicAdd(&agg[er[t][v] * DF + colN], x);
            }
    }
}

// ---------------------------------------------------------------------------
// Node kernel: one wave32 per 16-node tile.
//   a   = concat(h, agg)                       [16, 256]
//   t   = silu(a @ nW1 + nb1)                  [16, 128]
//   out = h + t @ nW2 + nb2                    [16, 128]   (no SiLU on layer 2)
// ---------------------------------------------------------------------------
__global__ __launch_bounds__(32)
void gcl_node_kernel(const float*  __restrict__ h,
                     const __bf16* __restrict__ hb,
                     const float*  __restrict__ agg,
                     const __bf16* __restrict__ W1p, const float* __restrict__ b1,
                     const __bf16* __restrict__ W2p, const float* __restrict__ b2,
                     float* __restrict__ out) {
    const int L    = threadIdx.x;
    const int r    = L & 15;
    const int half = L >> 4;
    const int n0   = blockIdx.x * 16;
    const int node = n0 + r;

    __shared__ __bf16 sE[16 * DF];

    v8f acc[8];
#pragma unroll
    for (int nt = 0; nt < 8; ++nt)
#pragma unroll
        for (int j = 0; j < 8; ++j) acc[nt][j] = 0.0f;

    // K 0..127: h side (already bf16)
#pragma unroll
    for (int kk = 0; kk < 4; ++kk) {
        const __bf16* src = hb + node * DF + kk * 32 + half * 8;
        v8bf lo = *(const v8bf*)(src);
        v8bf hi = *(const v8bf*)(src + 16);
        v16bf a;
#pragma unroll
        for (int i = 0; i < 8; ++i) { a[i] = lo[i]; a[8 + i] = hi[i]; }
#pragma unroll
        for (int nt = 0; nt < 8; ++nt) {
            const v16bf* bp = (const v16bf*)(W1p + ((kk * 8 + nt) << 9));
            v16bf b = bp[L];
            acc[nt] = __builtin_amdgcn_wmma_f32_16x16x32_bf16(
                false, a, false, b, (short)0, acc[nt], false, false);
        }
    }
    // K 128..255: agg side (fp32 -> bf16 on load)
#pragma unroll
    for (int kk = 4; kk < 8; ++kk) {
        const float* srcf = agg + node * DF + (kk - 4) * 32 + half * 8;
        v8f flo = *(const v8f*)(srcf);
        v8f fhi = *(const v8f*)(srcf + 16);
        v16bf a;
#pragma unroll
        for (int i = 0; i < 8; ++i) {
            a[i]     = (__bf16)flo[i];
            a[8 + i] = (__bf16)fhi[i];
        }
#pragma unroll
        for (int nt = 0; nt < 8; ++nt) {
            const v16bf* bp = (const v16bf*)(W1p + ((kk * 8 + nt) << 9));
            v16bf b = bp[L];
            acc[nt] = __builtin_amdgcn_wmma_f32_16x16x32_bf16(
                false, a, false, b, (short)0, acc[nt], false, false);
        }
    }

#pragma unroll
    for (int nt = 0; nt < 8; ++nt) {
        const int colN = nt * 16 + r;
        const float bias = b1[colN];
#pragma unroll
        for (int v = 0; v < 8; ++v) {
            float x = silu_f(acc[nt][v] + bias);
            sE[(v + 8 * half) * DF + colN] = (__bf16)x;
        }
    }
    __syncthreads();

    v8f acc2[8];
#pragma unroll
    for (int nt = 0; nt < 8; ++nt)
#pragma unroll
        for (int j = 0; j < 8; ++j) acc2[nt][j] = 0.0f;

#pragma unroll
    for (int kk = 0; kk < 4; ++kk) {
        const __bf16* src = sE + r * DF + kk * 32 + half * 8;
        v8bf lo = *(const v8bf*)(src);
        v8bf hi = *(const v8bf*)(src + 16);
        v16bf a;
#pragma unroll
        for (int i = 0; i < 8; ++i) { a[i] = lo[i]; a[8 + i] = hi[i]; }
#pragma unroll
        for (int nt = 0; nt < 8; ++nt) {
            const v16bf* bp = (const v16bf*)(W2p + ((kk * 8 + nt) << 9));
            v16bf b = bp[L];
            acc2[nt] = __builtin_amdgcn_wmma_f32_16x16x32_bf16(
                false, a, false, b, (short)0, acc2[nt], false, false);
        }
    }

    // bias + residual, store fp32 output (no SiLU on final layer)
#pragma unroll
    for (int nt = 0; nt < 8; ++nt) {
        const int colN = nt * 16 + r;
        const float bias = b2[colN];
#pragma unroll
        for (int v = 0; v < 8; ++v) {
            const int nodeM = n0 + v + 8 * half;
            const int o = nodeM * DF + colN;
            out[o] = h[o] + acc2[nt][v] + bias;
        }
    }
}

// ---------------------------------------------------------------------------
// Launch
// ---------------------------------------------------------------------------
extern "C" void kernel_launch(void* const* d_in, const int* in_sizes, int n_in,
                              void* d_out, int out_size, void* d_ws, size_t ws_size,
                              hipStream_t stream) {
    (void)in_sizes; (void)n_in; (void)out_size; (void)ws_size;

    const float* h   = (const float*)d_in[0];
    const int*   eix = (const int*)  d_in[1];
    const float* eW1 = (const float*)d_in[2];
    const float* eb1 = (const float*)d_in[3];
    const float* eW2 = (const float*)d_in[4];
    const float* eb2 = (const float*)d_in[5];
    const float* nW1 = (const float*)d_in[6];
    const float* nb1 = (const float*)d_in[7];
    const float* nW2 = (const float*)d_in[8];
    const float* nb2 = (const float*)d_in[9];

    // Workspace layout (all offsets 32B-aligned):
    //   agg   fp32 [N,128]   @ 0           25,600,000 B
    //   hb    bf16 [N,128]   @ 25,600,000  12,800,000 B
    //   eW1p  bf16 packed    @ 38,400,000      65,536 B
    //   eW2p  bf16 packed    @ 38,465,536      32,768 B
    //   nW1p  bf16 packed    @ 38,498,304      65,536 B
    //   nW2p  bf16 packed    @ 38,563,840      32,768 B
    char*   ws   = (char*)d_ws;
    float*  agg  = (float*) (ws);
    __bf16* hb   = (__bf16*)(ws + 25600000);
    __bf16* W1p  = (__bf16*)(ws + 38400000);
    __bf16* W2p  = (__bf16*)(ws + 38465536);
    __bf16* W3p  = (__bf16*)(ws + 38498304);
    __bf16* W4p  = (__bf16*)(ws + 38563840);

    const int* rows = eix;
    const int* cols = eix + N_EDGES;

    const int totH = N_NODES * DF;  // 6,400,000
    gcl_prep_h<<<(totH + 255) / 256, 256, 0, stream>>>(h, hb, agg, totH);
    gcl_pack_w<<<(32768 + 255) / 256, 256, 0, stream>>>(eW1, W1p, 8);
    gcl_pack_w<<<(16384 + 255) / 256, 256, 0, stream>>>(eW2, W2p, 4);
    gcl_pack_w<<<(32768 + 255) / 256, 256, 0, stream>>>(nW1, W3p, 8);
    gcl_pack_w<<<(16384 + 255) / 256, 256, 0, stream>>>(nW2, W4p, 4);

    gcl_edge_kernel<<<N_EDGES / 32, 32, 0, stream>>>(hb, rows, cols,
                                                     W1p, eb1, W2p, eb2, agg);
    gcl_node_kernel<<<N_NODES / 16, 32, 0, stream>>>(h, hb, agg,
                                                     W3p, nb1, W4p, nb2,
                                                     (float*)d_out);
}
